// MaskRoIPool_39908836114973
// MI455X (gfx1250) — compile-verified, matrix operand order checked by reading, not run on hardware
//
#include <hip/hip_runtime.h>
#include <math.h>

#define NC   256
#define NH   40
#define NW   40
#define NHW  (NH * NW)      // 1600
#define NROI 64
#define PHB  7
#define PWB  7
#define RSCALE (1.0f / 16.0f)

typedef __attribute__((ext_vector_type(2)))  float    v2f;
typedef __attribute__((ext_vector_type(8)))  float    v8f;
typedef __attribute__((ext_vector_type(16))) _Float16 v16h;
typedef __attribute__((vector_size(16)))     int      vsi4;   // b128 payload type

#if defined(__gfx1250__) && \
    __has_builtin(__builtin_amdgcn_global_load_async_to_lds_b128) && \
    __has_builtin(__builtin_amdgcn_s_wait_asynccnt)
#define USE_ASYNC_LDS 1
#endif

// ---------------------------------------------------------------------------
// Kernel 1: saliency channel-sum via WMMA (f32 exact), global max, mask map.
// One workgroup (512 threads = 16 waves). s = sum_c fm[c,:,:] computed as
// D = ones(16x4) x B(4x16) accumulated over 64 K-steps per 16-position tile:
// every row of D equals the per-column (per-position) channel sum.
// ---------------------------------------------------------------------------
__global__ __launch_bounds__(512) void saliency_kernel(
    const float* __restrict__ fm, float* __restrict__ mask_out) {
  __shared__ float s_s[NHW];     // channel sums (6.4 KB)
  __shared__ float s_red[512];
  const int tid = threadIdx.x;

#if __has_builtin(__builtin_amdgcn_wmma_f32_16x16x4_f32)
  {
    const int wave = tid >> 5;
    const int lane = tid & 31;
    const int col  = lane & 15;        // output position within tile (N index)
    const int hi   = lane >> 4;        // K sub-row select (0/1)
    v2f a; a.x = 1.0f; a.y = 1.0f;     // A = ones (16x4)
    for (int t = wave; t < NHW / 16; t += 16) {   // 100 tiles of 16 positions
      const int p0 = t * 16;
      const float* base = fm + p0 + col;
      v8f acc = {};
      for (int c0 = 0; c0 < NC; c0 += 4) {
        v2f b;                          // B[k][n] = fm[c0+k][p0+n]
        b.x = base[(c0 + hi)     * NHW];
        b.y = base[(c0 + 2 + hi) * NHW];
        acc = __builtin_amdgcn_wmma_f32_16x16x4_f32(
            false, a, false, b, (short)0, acc, false, false);
      }
      if (lane < 16) s_s[p0 + col] = acc[0];   // D row 0 = column sums
    }
  }
#elif __has_builtin(__builtin_amdgcn_wmma_f32_16x16x32_f16)
  {
    const int wave = tid >> 5;
    const int lane = tid & 31;
    const int col  = lane & 15;
    const int kb   = (lane >> 4) * 8;   // K base for this half-wave
    v16h a;
    for (int e = 0; e < 16; ++e) a[e] = (_Float16)1.0f;
    for (int t = wave; t < NHW / 16; t += 16) {
      const int p0 = t * 16;
      const float* base = fm + p0 + col;
      v8f acc = {};
      for (int c0 = 0; c0 < NC; c0 += 32) {
        v16h b;
        for (int e = 0; e < 16; ++e) {
          const int k = (e < 8) ? (kb + e) : (16 + kb + (e - 8));
          b[e] = (_Float16)base[(c0 + k) * NHW];
        }
        acc = __builtin_amdgcn_wmma_f32_16x16x32_f16(
            false, a, false, b, (short)0, acc, false, false);
      }
      if (lane < 16) s_s[p0 + col] = acc[0];
    }
  }
#else
  for (int p = tid; p < NHW; p += 512) {
    float acc = 0.0f;
    for (int c = 0; c < NC; ++c) acc += fm[c * NHW + p];
    s_s[p] = acc;
  }
#endif
  __syncthreads();

  // Block-wide max of s
  float m = -INFINITY;
  for (int p = tid; p < NHW; p += 512) m = fmaxf(m, s_s[p]);
  s_red[tid] = m;
  __syncthreads();
  for (int off = 256; off > 0; off >>= 1) {
    if (tid < off) s_red[tid] = fmaxf(s_red[tid], s_red[tid + off]);
    __syncthreads();
  }
  const float inv = 1.0f / s_red[0];

  // mask_map = 0.5 + 0.4 * (s/smax)^4
  for (int p = tid; p < NHW; p += 512) {
    const float r  = s_s[p] * inv;
    const float r2 = r * r;
    mask_out[p] = 0.5f + 0.4f * (r2 * r2);
  }
}

// ---------------------------------------------------------------------------
// Kernel 2: masked adaptive max pooling. One block per (roi, 32-channel group).
// Mask (6.4 KB, shared by all channels of the block) is staged into LDS via
// async global->LDS copies (ASYNCcnt path) overlapping the RoI box setup.
// ---------------------------------------------------------------------------
__device__ __forceinline__ int zoom1(float v, int limit) {
  int r = (int)rintf(v * RSCALE);     // RNE, matches jnp.round
  return min(r, limit - 1);           // reference clamps only the upper side
}

__global__ __launch_bounds__(256) void pool_kernel(
    const float* __restrict__ fm, const float* __restrict__ rois1,
    const float* __restrict__ rois2, const float* __restrict__ mask,
    float* __restrict__ out) {
  __shared__ __attribute__((aligned(16))) float s_mask[NHW];

#if USE_ASYNC_LDS
  {
    for (int p = threadIdx.x; p < NHW / 4; p += 256) {
      __builtin_amdgcn_global_load_async_to_lds_b128(
          (vsi4 __attribute__((address_space(1)))*)(mask + 4 * p),
          (vsi4 __attribute__((address_space(3)))*)(s_mask + 4 * p),
          0, 0);
    }
  }
#else
  for (int p = threadIdx.x; p < NHW; p += 256) s_mask[p] = mask[p];
#endif

  const int n  = blockIdx.x;
  const int cg = blockIdx.y;          // 32 channels per group

  const float* r1 = rois1 + n * 5;
  const float* r2 = rois2 + n * 5;
  const int x1a = zoom1(r1[1], NW), y1a = zoom1(r1[2], NH);
  const int x2a = zoom1(r1[3], NW), y2a = zoom1(r1[4], NH);
  const int x1b = zoom1(r2[1], NW), y1b = zoom1(r2[2], NH);
  const int x2b = zoom1(r2[3], NW), y2b = zoom1(r2[4], NH);
  const int ux1 = min(x1a, x1b), uy1 = min(y1a, y1b);
  const int ux2 = max(x2a, x2b), uy2 = max(y2a, y2b);
  const int hb = uy2 - uy1 + 1;
  const int wb = ux2 - ux1 + 1;

#if USE_ASYNC_LDS
  __builtin_amdgcn_s_wait_asynccnt(0);   // my wave's LDS writes landed
#endif
  __syncthreads();                       // all waves' staging visible

  for (int idx = threadIdx.x; idx < 32 * PHB * PWB; idx += 256) {
    const int cl  = idx / (PHB * PWB);
    const int bin = idx - cl * (PHB * PWB);
    const int i = bin / PWB;
    const int j = bin - i * PWB;
    const int c = cg * 32 + cl;

    // Adaptive bin span [start, end) relative to union box origin (inclusive hi)
    const int ys0 = uy1 + (i * hb) / PHB;
    const int ys1 = uy1 + ((i + 1) * hb + PHB - 1) / PHB;
    const int xs0 = ux1 + (j * wb) / PWB;
    const int xs1 = ux1 + ((j + 1) * wb + PWB - 1) / PWB;

    const float* f = fm + c * NHW;
    __builtin_prefetch(f + ys0 * NW + xs0, 0, 1);   // global_prefetch_b8

    float mx = -INFINITY;
    for (int y = ys0; y < ys1; ++y) {
      const bool iya = (y >= y1a) & (y <= y2a);
      const bool iyb = (y >= y1b) & (y <= y2b);
      const int row = y * NW;
      for (int x = xs0; x < xs1; ++x) {
        const bool ov = (iya & (x >= x1a) & (x <= x2a)) |
                        (iyb & (x >= x1b) & (x <= x2b));
        const float eff = ov ? 1.0f : s_mask[row + x];
        mx = fmaxf(mx, f[row + x] * eff);
      }
    }
    out[((n * NC + c) * PHB + i) * PWB + j] = mx;
  }
}

// ---------------------------------------------------------------------------
extern "C" void kernel_launch(void* const* d_in, const int* in_sizes, int n_in,
                              void* d_out, int out_size, void* d_ws, size_t ws_size,
                              hipStream_t stream) {
  const float* fm = (const float*)d_in[0];   // (1,256,40,40) f32
  const float* r1 = (const float*)d_in[1];   // (64,5) f32
  const float* r2 = (const float*)d_in[2];   // (64,5) f32
  float* out  = (float*)d_out;               // (64,256,7,7) f32
  float* mask = (float*)d_ws;                // 1600 f32 scratch

  saliency_kernel<<<dim3(1), dim3(512), 0, stream>>>(fm, mask);
  pool_kernel<<<dim3(NROI, NC / 32), dim3(256), 0, stream>>>(fm, r1, r2, mask, out);
}
